// AttFusion_41394894799046
// MI455X (gfx1250) — matrix operand order repaired and checked by reading, not compiled, not run on hardware
//
#include <hip/hip_runtime.h>

// AttFusion for MI455X (gfx1250, wave32, WMMA f16).
// x:(14,256,100,252) f32, 4 ragged groups (record_len=[5,3,4,2]); only the
// ego frame (l=0) of each group survives, so q/score are needed for row 0 only.
//
// Blocking: each block = one 16-position spatial tile x TWO groups.  All
// groups share Wq/Wk/Wv, so fusing two groups doubles the A-rows amortizing
// every register-resident B fragment -> halves weight re-read traffic from L2
// (the dominant byte stream: ~2.4 GB L2 vs ~0.46 GB HBM for x/out).
// Pipeline: stage x tile -> LDS f16 -> v_wmma_f32_16x16x32_f16 projections
// (q only for the 2 ego frames) -> score via cross-lane reduce of q*k frags
// -> softmax -> ctx from LDS v -> out = gamma*ctx + v[ego].

#define CDIM   256
#define SDIM   25200      // W*H = 100*252 = 16 * 1575 (exact tiling)
#define STILE  16
#define MAXF   8          // max frames per block (two groups: 5+3)
#define NPAIR  2          // 4 groups -> 2 pairs

typedef _Float16 v16h __attribute__((ext_vector_type(16)));
typedef _Float16 h2   __attribute__((ext_vector_type(2)));
typedef float    v8f  __attribute__((ext_vector_type(8)));

__global__ __launch_bounds__(256, 1)
void attfusion_kernel(const float* __restrict__ x,
                      const float* __restrict__ Wq, const float* __restrict__ bq,
                      const float* __restrict__ Wk, const float* __restrict__ bk,
                      const float* __restrict__ Wv, const float* __restrict__ bv,
                      const float* __restrict__ gamma_p,
                      const int*   __restrict__ record_len,
                      float*       __restrict__ out)
{
    __shared__ _Float16 sA[MAXF][STILE][CDIM];   // staged x, f16 (64 KB)
    __shared__ float    sV[MAXF][STILE][CDIM];   // v projection, f32 (128 KB)
    __shared__ float    sScore[2][STILE][MAXF];  // per-group q0.k_m sums (1 KB)

    const int tid  = threadIdx.x;
    const int lane = tid & 31;
    const int wave = tid >> 5;
    const int pair = blockIdx.y;           // groups {2p, 2p+1}
    const int s0   = blockIdx.x * STILE;   // spatial tile start

    const int g0 = 2 * pair;
    int base0 = 0;
    for (int i = 0; i < g0; ++i) base0 += record_len[i];
    int L0 = record_len[g0];
    int L1 = record_len[g0 + 1];
    const int base1 = base0 + L0;
    int LT = L0 + L1;
    if (LT > MAXF) { LT = MAXF; L1 = LT - L0; }

    // ---- Phase A: stage x tile -> LDS f16 (float4 loads); zero scores ----
    {
        const int total4 = LT * CDIM * 4;          // (frame, channel, m-quad)
        for (int i = tid; i < total4; i += 256) {
            int q  = i & 3;
            int rc = i >> 2;
            int c  = rc & (CDIM - 1);
            int l  = rc >> 8;
            int frame = (l < L0) ? (base0 + l) : (base1 + (l - L0));
            const float4 xv = *(const float4*)
                &x[((size_t)frame * CDIM + c) * (size_t)SDIM + (size_t)(s0 + 4 * q)];
            int m = 4 * q;
            sA[l][m + 0][c] = (_Float16)xv.x;
            sA[l][m + 1][c] = (_Float16)xv.y;
            sA[l][m + 2][c] = (_Float16)xv.z;
            sA[l][m + 3][c] = (_Float16)xv.w;
        }
        ((float*)sScore)[tid] = 0.0f;              // 2*16*8 == 256
    }
    __syncthreads();

    // ---- Phase B: WMMA projections; score accumulated in-register ----
    const int mN   = lane & 15;   // A-frag M row == B/C/D-frag N column
    const int half = lane >> 4;

    for (int nt = wave; nt < 16; nt += 8) {        // 16 N-tiles of channels
        v8f qa0 = {0.f,0.f,0.f,0.f,0.f,0.f,0.f,0.f};
        v8f qa1 = qa0;
        v8f ka[MAXF], va[MAXF];
        #pragma unroll
        for (int l = 0; l < MAXF; ++l) { ka[l] = qa0; va[l] = qa0; }

        const int row = nt * 16 + mN;              // output-channel row of W

        for (int kk = 0; kk < CDIM; kk += 32) {
            // B fragments: B[k][n] = W[n][k] -> contiguous along W rows.
            v16h bqf, bkf, bvf;
            #pragma unroll
            for (int j = 0; j < 8; ++j) {
                int k0 = kk + ((j < 4) ? (2*j + 8*half) : (16 + 2*(j-4) + 8*half));
                float2 wq2 = *(const float2*)&Wq[(size_t)row * CDIM + k0];
                float2 wk2 = *(const float2*)&Wk[(size_t)row * CDIM + k0];
                float2 wv2 = *(const float2*)&Wv[(size_t)row * CDIM + k0];
                bqf[2*j] = (_Float16)wq2.x; bqf[2*j+1] = (_Float16)wq2.y;
                bkf[2*j] = (_Float16)wk2.x; bkf[2*j+1] = (_Float16)wk2.y;
                bvf[2*j] = (_Float16)wv2.x; bvf[2*j+1] = (_Float16)wv2.y;
            }
            #pragma unroll
            for (int l = 0; l < MAXF; ++l) {
                if (l < LT) {   // uniform across wave -> EXEC stays all-ones
                    v16h a;
                    #pragma unroll
                    for (int j = 0; j < 8; ++j) {
                        int kb = (j < 4) ? (2*j + 8*half) : (16 + 2*(j-4) + 8*half);
                        h2 p = *(const h2*)&sA[l][mN][kk + kb];
                        a[2*j] = p.x; a[2*j+1] = p.y;
                    }
                    if (l == 0)   // ego of group 0
                        qa0 = __builtin_amdgcn_wmma_f32_16x16x32_f16(
                                  false, a, false, bqf, (short)0, qa0, false, false);
                    if (l == L0)  // ego of group 1
                        qa1 = __builtin_amdgcn_wmma_f32_16x16x32_f16(
                                  false, a, false, bqf, (short)0, qa1, false, false);
                    ka[l] = __builtin_amdgcn_wmma_f32_16x16x32_f16(
                                  false, a, false, bkf, (short)0, ka[l], false, false);
                    va[l] = __builtin_amdgcn_wmma_f32_16x16x32_f16(
                                  false, a, false, bvf, (short)0, va[l], false, false);
                }
            }
        }

        // Bias, per-frame score partials (q.k over this 16-channel slab), v out.
        const float bqv = bq[row], bkv = bk[row], bvv = bv[row];
        #pragma unroll
        for (int r = 0; r < 8; ++r) { qa0[r] += bqv; qa1[r] += bqv; }

        #pragma unroll
        for (int l = 0; l < MAXF; ++l) {
            if (l < LT) {
                const int grp = (l >= L0) ? 1 : 0;
                const int lm  = l - (grp ? L0 : 0);
                #pragma unroll
                for (int r = 0; r < 8; ++r) {
                    ka[l][r] += bkv;
                    va[l][r] += bvv;
                    float qv = grp ? qa1[r] : qa0[r];
                    float pp = qv * ka[l][r];          // reduce over 16 N-lanes
                    pp += __shfl_xor(pp, 1, 32);
                    pp += __shfl_xor(pp, 2, 32);
                    pp += __shfl_xor(pp, 4, 32);
                    pp += __shfl_xor(pp, 8, 32);
                    if (mN == 0) atomicAdd(&sScore[grp][r + 8 * half][lm], pp);
                    sV[l][r + 8 * half][row] = va[l][r];
                }
            }
        }
    }
    __syncthreads();

    // ---- Phase C: softmax over frames, per group, per spatial position ----
    if (tid < 2 * STILE) {
        const int grp = tid >> 4;
        const int s   = tid & 15;
        const int Lg  = grp ? L1 : L0;
        float sc[MAXF];
        float mx = -3.0e38f;
        for (int l = 0; l < MAXF; ++l)
            if (l < Lg) { sc[l] = sScore[grp][s][l] * 0.0625f; mx = fmaxf(mx, sc[l]); }
        float sum = 0.f;
        for (int l = 0; l < MAXF; ++l)
            if (l < Lg) { sc[l] = __expf(sc[l] - mx); sum += sc[l]; }
        const float inv = 1.f / sum;
        for (int l = 0; l < MAXF; ++l)
            if (l < Lg) sScore[grp][s][l] = sc[l] * inv;
    }
    __syncthreads();

    // ---- Phase D: ctx = attn @ v ; out = gamma*ctx + v[ego], both groups ----
    const float gm = gamma_p[0];
    for (int idx = tid; idx < 2 * STILE * CDIM; idx += 256) {
        const int grp = idx >> 12;
        const int r   = idx & 4095;
        const int c   = r >> 4;
        const int s   = r & 15;
        const int fo  = grp ? L0 : 0;     // first frame of this group
        const int Lg  = grp ? L1 : L0;
        float ctx = 0.f;
        #pragma unroll
        for (int l = 0; l < MAXF; ++l)
            if (l < Lg) ctx += sScore[grp][s][l] * sV[fo + l][s][c];
        const int gout = 2 * pair + grp;
        out[((size_t)(gout * CDIM) + c) * (size_t)SDIM + (size_t)(s0 + s)] =
            gm * ctx + sV[fo][s][c];
    }
}

extern "C" void kernel_launch(void* const* d_in, const int* in_sizes, int n_in,
                              void* d_out, int out_size, void* d_ws, size_t ws_size,
                              hipStream_t stream) {
    const float* x     = (const float*)d_in[0];
    const float* Wq    = (const float*)d_in[1];
    const float* bq    = (const float*)d_in[2];
    const float* Wk    = (const float*)d_in[3];
    const float* bk    = (const float*)d_in[4];
    const float* Wv    = (const float*)d_in[5];
    const float* bv    = (const float*)d_in[6];
    const float* gamma = (const float*)d_in[7];
    const int*   rl    = (const int*)d_in[8];
    float*       out   = (float*)d_out;

    dim3 grid(SDIM / STILE, NPAIR);    // 1575 x 2 blocks
    dim3 block(256);                   // 8 waves (wave32)
    attfusion_kernel<<<grid, block, 0, stream>>>(x, Wq, bq, Wk, bk, Wv, bv,
                                                 gamma, rl, out);
}
